// DecoderOnlyTransformer_25598005084378
// MI455X (gfx1250) — matrix operand order repaired and checked by reading, not compiled, not run on hardware
//
#include <hip/hip_runtime.h>
#include <cstdint>
#include <cmath>

// ---------------------------------------------------------------------------
// CDNA5 (gfx1250) decoder-only transformer forward.
// Only the LAST layer matters (reference bug reproduced faithfully).
// All GEMMs use v_wmma_f32_16x16x32_bf16 (wave32), fp32 accumulate.
// 64x64 wave tile (16 WMMAs / K-step, 32 FLOP/byte of fragment traffic),
// single live A-fragment to stay under the 256-VGPR window (no spills).
// Output streamed with nontemporal stores so Wout^T (131 MB bf16) owns L2.
// ---------------------------------------------------------------------------

typedef __bf16 bf16_t;
typedef bf16_t v16bf __attribute__((ext_vector_type(16)));
typedef bf16_t v8bf  __attribute__((ext_vector_type(8)));
typedef float  v8f   __attribute__((ext_vector_type(8)));

#define SEQ    2048
#define DMODEL 1024
#define DK     64
#define NH     16
#define ROWS   4096   // B * S

enum { ST_BF16 = 0, ST_F32 = 1, ST_BF16_TANH = 2, ST_VT = 3 };

__device__ __forceinline__ bf16_t f2bf(float f) {
    unsigned u = __builtin_bit_cast(unsigned, f);
    u += 0x7fffu + ((u >> 16) & 1u);            // round-to-nearest-even
    unsigned short s = (unsigned short)(u >> 16);
    return __builtin_bit_cast(bf16_t, s);
}

__device__ __forceinline__ v8f wmma_bf16(v16bf a, v16bf b, v8f c) {
    return __builtin_amdgcn_wmma_f32_16x16x32_bf16(
        /*neg_a=*/false, a, /*neg_b=*/false, b,
        /*c_mod=*/(short)0, c, /*reuse_a=*/false, /*reuse_b=*/false);
}

// A-matrix fragment (16x32 bf16): lane (m = lane&15, lh = lane>>4) holds
// K = lh*16 + [0..7]  and  K = lh*16 + 8 + [0..7]   (two 16-byte chunks)
__device__ __forceinline__ v16bf load_a_pair(const bf16_t* __restrict__ p, int lh) {
    v8bf c0 = *reinterpret_cast<const v8bf*>(p + lh * 16);
    v8bf c1 = *reinterpret_cast<const v8bf*>(p + lh * 16 + 8);
    v16bf r;
#pragma unroll
    for (int i = 0; i < 8; ++i) { r[i] = c0[i]; r[i + 8] = c1[i]; }
    return r;
}

// ---------------------------------------------------------------------------
// Embedding + sinusoidal positional encoding -> bf16 h[ROWS, DMODEL]
// ---------------------------------------------------------------------------
__global__ void embed_kernel(const int* __restrict__ x, const float* __restrict__ E,
                             bf16_t* __restrict__ hB) {
    int idx = blockIdx.x * blockDim.x + threadIdx.x;   // ROWS*DMODEL threads
    int d   = idx & (DMODEL - 1);
    int row = idx >> 10;
    int s   = row & (SEQ - 1);
    int tok = x[row];
    float freq = __expf(-(float)(d & ~1) * (9.210340371976184f / (float)DMODEL));
    float ang  = (float)s * freq;
    float pe   = (d & 1) ? __cosf(ang) : __sinf(ang);
    hB[idx] = f2bf(E[(size_t)tok * DMODEL + d] + pe);
}

// ---------------------------------------------------------------------------
// LDS-tiled transpose-convert fp32 W[K,N] -> bf16 Wt[N,K].
// Coalesced NT fp32 reads (one-shot data, keep out of L2), coalesced bf16
// writes. block = (32,8), grid = (N/32, K/32).
// ---------------------------------------------------------------------------
__global__ __launch_bounds__(256)
void wtrans_kernel(const float* __restrict__ W, bf16_t* __restrict__ Wt,
                   int K, int N) {
    __shared__ bf16_t tile[32][33];
    const int n0 = blockIdx.x * 32;
    const int k0 = blockIdx.y * 32;
    const int tx = threadIdx.x;      // 0..31
    const int ty = threadIdx.y;      // 0..7
#pragma unroll
    for (int i = 0; i < 32; i += 8)
        tile[ty + i][tx] =
            f2bf(__builtin_nontemporal_load(&W[(size_t)(k0 + ty + i) * N + n0 + tx]));
    __syncthreads();
#pragma unroll
    for (int i = 0; i < 32; i += 8)
        Wt[(size_t)(n0 + ty + i) * K + k0 + tx] = tile[tx][ty + i];
}

// ---------------------------------------------------------------------------
// bf16 WMMA GEMM: C[M,N] = act(A[M,K] * Bt[N,K]^T + bias)
// block = 128 threads (4 waves, 2x2), block tile 128x128, wave tile 64x64.
// 16 WMMAs per K-step of 32; only one A fragment live at a time.
// MODE selects epilogue.
// ---------------------------------------------------------------------------
template <int MODE>
__global__ __launch_bounds__(128)
void gemm_kernel(const bf16_t* __restrict__ A, const bf16_t* __restrict__ Bt,
                 const float* __restrict__ bias, void* __restrict__ Cout,
                 int M, int N, int K) {
    const int lane = threadIdx.x & 31;
    const int wid  = threadIdx.x >> 5;
    const int ln   = lane & 15;
    const int lh   = lane >> 4;
    const int m0   = blockIdx.x * 128 + (wid & 1) * 64;
    const int n0   = blockIdx.y * 128 + (wid >> 1) * 64;

    const bf16_t* ap[4];
    const bf16_t* bp[4];
#pragma unroll
    for (int i = 0; i < 4; ++i) {
        ap[i] = A  + (size_t)(m0 + i * 16 + ln) * K;
        bp[i] = Bt + (size_t)(n0 + i * 16 + ln) * K;
    }

    v8f acc[4][4];
#pragma unroll
    for (int i = 0; i < 4; ++i)
#pragma unroll
        for (int j = 0; j < 4; ++j) acc[i][j] = {};

    for (int kk = 0; kk < K; kk += 32) {
        __builtin_prefetch(ap[0] + kk + 64, 0, 3);   // global_prefetch next panels
        __builtin_prefetch(bp[0] + kk + 64, 0, 3);
        v16bf bfr[4];
#pragma unroll
        for (int j = 0; j < 4; ++j)
            bfr[j] = *reinterpret_cast<const v16bf*>(bp[j] + kk + lh * 16);
#pragma unroll
        for (int i = 0; i < 4; ++i) {
            const v16bf af = load_a_pair(ap[i] + kk, lh);   // one live A frag
#pragma unroll
            for (int j = 0; j < 4; ++j)
                acc[i][j] = wmma_bf16(af, bfr[j], acc[i][j]);
        }
    }

    float bs[4];
#pragma unroll
    for (int j = 0; j < 4; ++j) bs[j] = bias[n0 + j * 16 + ln];

#pragma unroll
    for (int i = 0; i < 4; ++i) {
#pragma unroll
        for (int r = 0; r < 8; ++r) {
            const int row = m0 + i * 16 + lh * 8 + r;
            float*  crowf = (float*)Cout  + (size_t)row * N + n0 + ln;
            bf16_t* crowb = (bf16_t*)Cout + (size_t)row * N + n0 + ln;
#pragma unroll
            for (int j = 0; j < 4; ++j) {
                float v = acc[i][j][r] + bs[j];
                if (MODE == ST_F32) {
                    // stream fp32 output past L2 (keep Wout^T resident)
                    __builtin_nontemporal_store(v, crowf + j * 16);
                } else if (MODE == ST_BF16_TANH) {
                    crowb[j * 16] = f2bf(tanhf(v));
                } else if (MODE == ST_VT) {
                    // fused V head-transpose: Vt[b, h, d, s] (contiguous-s)
                    const int b = row >> 11, s = row & (SEQ - 1);
                    const int col = n0 + j * 16 + ln;
                    const int h = col >> 6,  dd = col & (DK - 1);
                    ((bf16_t*)Cout)[((size_t)((b * NH + h) * DK + dd)) * SEQ + s] = f2bf(v);
                } else {
                    crowb[j * 16] = f2bf(v);
                }
            }
        }
    }
}

// ---------------------------------------------------------------------------
// Causal flash attention. One wave (blockDim=32) per 16 query rows.
// grid = B * NH * (SEQ/16) = 4096 blocks.
// Q,K bf16 [ROWS, DMODEL] (head cols h*64..), Vt bf16 [b,h,DK,SEQ].
// ---------------------------------------------------------------------------
__global__ __launch_bounds__(32)
void attn_kernel(const bf16_t* __restrict__ Qb, const bf16_t* __restrict__ Kb,
                 const bf16_t* __restrict__ Vt, bf16_t* __restrict__ attB) {
    __shared__ bf16_t Pl[16 * 32];

    const int lane = threadIdx.x & 31;
    const int ln   = lane & 15;
    const int lh   = lane >> 4;
    const int bid  = blockIdx.x;
    const int qblk = bid & 127;           // SEQ/16 = 128
    const int hd   = (bid >> 7) & (NH - 1);
    const int bb   = bid >> 11;
    const int qr0  = qblk * 16;

    const bf16_t* qrow = Qb + ((size_t)(bb * SEQ + qr0 + ln)) * DMODEL + hd * DK;
    const v16bf aq0 = load_a_pair(qrow, lh);        // Kdim 0..31
    const v16bf aq1 = load_a_pair(qrow + 32, lh);   // Kdim 32..63

    const bf16_t* kbase = Kb + (size_t)bb * SEQ * DMODEL + hd * DK;
    const bf16_t* vtb   = Vt + (size_t)(bb * NH + hd) * DK * SEQ;

    float mr[8], lr[8];
    v8f o[4] = {};
#pragma unroll
    for (int r = 0; r < 8; ++r) { mr[r] = -3.0e38f; lr[r] = 0.0f; }

    const int nj = qr0 / 32 + 1;
    for (int j = 0; j < nj; ++j) {
        const int kv0 = j * 32;

        // ----- scores: two 16x16 tiles (kv cols 0..15 and 16..31) -----
        v8f s0 = {}, s1 = {};
        {
            const bf16_t* kr0 = kbase + (size_t)(kv0 + ln) * DMODEL;
            const bf16_t* kr1 = kbase + (size_t)(kv0 + 16 + ln) * DMODEL;
            v16bf bk;
            bk = *reinterpret_cast<const v16bf*>(kr0 + lh * 16);      s0 = wmma_bf16(aq0, bk, s0);
            bk = *reinterpret_cast<const v16bf*>(kr0 + 32 + lh * 16); s0 = wmma_bf16(aq1, bk, s0);
            bk = *reinterpret_cast<const v16bf*>(kr1 + lh * 16);      s1 = wmma_bf16(aq0, bk, s1);
            bk = *reinterpret_cast<const v16bf*>(kr1 + 32 + lh * 16); s1 = wmma_bf16(aq1, bk, s1);
        }

        // ----- online softmax update (per C-layout row r) -----
#pragma unroll
        for (int r = 0; r < 8; ++r) {
            const int qg = qr0 + lh * 8 + r;
            float x0 = (kv0 + ln      > qg) ? -3.0e38f : s0[r] * 0.125f;
            float x1 = (kv0 + 16 + ln > qg) ? -3.0e38f : s1[r] * 0.125f;
            float mj = fmaxf(x0, x1);
            mj = fmaxf(mj, __shfl_xor(mj, 1));
            mj = fmaxf(mj, __shfl_xor(mj, 2));
            mj = fmaxf(mj, __shfl_xor(mj, 4));
            mj = fmaxf(mj, __shfl_xor(mj, 8));
            const float mn = fmaxf(mr[r], mj);
            const float al = __expf(mr[r] - mn);
            mr[r] = mn;
            const float p0 = __expf(x0 - mn);
            const float p1 = __expf(x1 - mn);
            float rs = p0 + p1;
            rs += __shfl_xor(rs, 1);
            rs += __shfl_xor(rs, 2);
            rs += __shfl_xor(rs, 4);
            rs += __shfl_xor(rs, 8);
            lr[r] = lr[r] * al + rs;
#pragma unroll
            for (int t = 0; t < 4; ++t) o[t][r] *= al;
            Pl[(lh * 8 + r) * 32 + ln]      = f2bf(p0);
            Pl[(lh * 8 + r) * 32 + 16 + ln] = f2bf(p1);
        }
        __syncthreads();   // P (C-layout) visible for A-layout reload

        // ----- P (16x32) x V (32x64): 4 WMMAs -----
        v8bf pc0 = *reinterpret_cast<const v8bf*>(&Pl[ln * 32 + lh * 16]);
        v8bf pc1 = *reinterpret_cast<const v8bf*>(&Pl[ln * 32 + lh * 16 + 8]);
        v16bf ap;
#pragma unroll
        for (int i = 0; i < 8; ++i) { ap[i] = pc0[i]; ap[i + 8] = pc1[i]; }

#pragma unroll
        for (int t = 0; t < 4; ++t) {
            v16bf bv = *reinterpret_cast<const v16bf*>(
                vtb + (size_t)(t * 16 + ln) * SEQ + kv0 + lh * 16);
            o[t] = wmma_bf16(ap, bv, o[t]);
        }
        __syncthreads();   // protect Pl before next iteration overwrites it
    }

    // ----- normalize and store att (bf16, [ROWS, DMODEL]) -----
#pragma unroll
    for (int r = 0; r < 8; ++r) {
        const float inv = 1.0f / lr[r];
        const size_t rowoff = (size_t)(bb * SEQ + qr0 + lh * 8 + r) * DMODEL + hd * DK;
#pragma unroll
        for (int t = 0; t < 4; ++t)
            attB[rowoff + t * 16 + ln] = f2bf(o[t][r] * inv);
    }
}

// ---------------------------------------------------------------------------
// Host-side launch
// ---------------------------------------------------------------------------
extern "C" void kernel_launch(void* const* d_in, const int* in_sizes, int n_in,
                              void* d_out, int out_size, void* d_ws, size_t ws_size,
                              hipStream_t stream) {
    const int*   x    = (const int*)  d_in[0];
    const float* E    = (const float*)d_in[1];
    const float* Wq   = (const float*)d_in[2];
    const float* bq   = (const float*)d_in[3];
    const float* Wk   = (const float*)d_in[4];
    const float* bk   = (const float*)d_in[5];
    const float* Wv   = (const float*)d_in[6];
    const float* bv   = (const float*)d_in[7];
    const float* Wo   = (const float*)d_in[8];
    const float* bo   = (const float*)d_in[9];
    const float* W1   = (const float*)d_in[10];
    const float* b1   = (const float*)d_in[11];
    const float* Wout = (const float*)d_in[12];
    const float* bout = (const float*)d_in[13];

    const int D = DMODEL, M = 4096, OMEGA = 16000;
    const int Lnum = in_sizes[2] / (D * D);
    const int ll   = Lnum - 1;   // only the last layer matters (reference bug)

    const float* Wq_l = Wq + (size_t)ll * D * D;
    const float* Wk_l = Wk + (size_t)ll * D * D;
    const float* Wv_l = Wv + (size_t)ll * D * D;
    const float* Wo_l = Wo + (size_t)ll * D * D;
    const float* W1_l = W1 + (size_t)ll * D * M;
    const float* bq_l = bq + (size_t)ll * D;
    const float* bk_l = bk + (size_t)ll * D;
    const float* bv_l = bv + (size_t)ll * D;
    const float* bo_l = bo + (size_t)ll * D;
    const float* b1_l = b1 + (size_t)ll * M;

    // ---- carve workspace (bf16 buffers, 256B aligned) ----
    char* p = (char*)d_ws;
    auto carve = [&](size_t bytes) -> void* {
        void* r = (void*)p;
        p += (bytes + 255) & ~(size_t)255;
        return r;
    };
    bf16_t* hB    = (bf16_t*)carve((size_t)ROWS * D * 2);
    bf16_t* WqT   = (bf16_t*)carve((size_t)D * D * 2);
    bf16_t* WkT   = (bf16_t*)carve((size_t)D * D * 2);
    bf16_t* WvT   = (bf16_t*)carve((size_t)D * D * 2);
    bf16_t* WoT   = (bf16_t*)carve((size_t)D * D * 2);
    bf16_t* W1T   = (bf16_t*)carve((size_t)D * M * 2);
    bf16_t* WoutT = (bf16_t*)carve((size_t)M * OMEGA * 2);
    bf16_t* QB    = (bf16_t*)carve((size_t)ROWS * D * 2);
    bf16_t* KB    = (bf16_t*)carve((size_t)ROWS * D * 2);
    bf16_t* VtB   = (bf16_t*)carve((size_t)ROWS * D * 2);
    bf16_t* attB  = (bf16_t*)carve((size_t)ROWS * D * 2);
    bf16_t* aoB   = (bf16_t*)carve((size_t)ROWS * D * 2);
    bf16_t* ffnB  = (bf16_t*)carve((size_t)ROWS * M * 2);

    const dim3 th128(128);
    const dim3 thT(32, 8);

    // 1) embedding + positional encoding
    embed_kernel<<<ROWS * D / 256, 256, 0, stream>>>(x, E, hB);

    // 2) weight transpose-converts (fp32 -> bf16, W^T), LDS-tiled
    wtrans_kernel<<<dim3(D / 32, D / 32), thT, 0, stream>>>(Wq_l, WqT, D, D);
    wtrans_kernel<<<dim3(D / 32, D / 32), thT, 0, stream>>>(Wk_l, WkT, D, D);
    wtrans_kernel<<<dim3(D / 32, D / 32), thT, 0, stream>>>(Wv_l, WvT, D, D);
    wtrans_kernel<<<dim3(D / 32, D / 32), thT, 0, stream>>>(Wo_l, WoT, D, D);
    wtrans_kernel<<<dim3(M / 32, D / 32), thT, 0, stream>>>(W1_l, W1T, D, M);
    wtrans_kernel<<<dim3(OMEGA / 32, M / 32), thT, 0, stream>>>(Wout, WoutT, M, OMEGA);

    // 3) Q, K projections; V projection fused with head-transpose
    gemm_kernel<ST_BF16><<<dim3(ROWS / 128, D / 128), th128, 0, stream>>>(hB, WqT, bq_l, QB, ROWS, D, D);
    gemm_kernel<ST_BF16><<<dim3(ROWS / 128, D / 128), th128, 0, stream>>>(hB, WkT, bk_l, KB, ROWS, D, D);
    gemm_kernel<ST_VT  ><<<dim3(ROWS / 128, D / 128), th128, 0, stream>>>(hB, WvT, bv_l, VtB, ROWS, D, D);

    // 4) causal flash attention
    attn_kernel<<<2 * NH * (SEQ / 16), 32, 0, stream>>>(QB, KB, VtB, attB);

    // 5) output projection, FFN (tanh), vocab projection (fp32, NT stores)
    gemm_kernel<ST_BF16     ><<<dim3(ROWS / 128, D / 128),     th128, 0, stream>>>(attB, WoT, bo_l, aoB, ROWS, D, D);
    gemm_kernel<ST_BF16_TANH><<<dim3(ROWS / 128, M / 128),     th128, 0, stream>>>(aoB, W1T, b1_l, ffnB, ROWS, M, D);
    gemm_kernel<ST_F32      ><<<dim3(ROWS / 128, OMEGA / 128), th128, 0, stream>>>(ffnB, WoutT, bout, d_out, ROWS, OMEGA, M);
}
